// MarketGNN_5497558139445
// MI455X (gfx1250) — matrix operand order repaired and checked by reading, not compiled, not run on hardware
//
#include <hip/hip_runtime.h>
#include <hip/hip_bf16.h>
#include <math.h>

// ---------------------------------------------------------------------------
// MarketGNN (3-layer GAT) forward for gfx1250 (MI455X).
// Dense projections: bf16 WMMA (v_wmma_f32_16x16x32_bf16). The entire 64-col
// B panel (<=64KB) is staged in LDS once per block, so the K loop is pure
// global(A) + ds(B) + wmma with no barriers. Wave tile = 32x64 (8 acc).
// Edge softmax/aggregation via f32 atomics; BN/LN/pool/MLP as light kernels.
// ---------------------------------------------------------------------------

typedef __bf16 bf16_t;
typedef __attribute__((ext_vector_type(16))) __bf16 bf16x16;
typedef __attribute__((ext_vector_type(8)))  __bf16 bf16x8;
typedef __attribute__((ext_vector_type(8)))  float  f32x8;

#define NN    30000
#define EE    480000
#define ETOT  (EE + NN)
#define GG    30
#define DIN   256
#define DHH   64
#define HH    8
#define C512  512
#define KMAX  512

// ---------------------------------------------------------------------------
// fp32 -> bf16 conversion (row-major copy)
__global__ __launch_bounds__(256)
void k_f2bf(const float* __restrict__ in, bf16_t* __restrict__ out, int n) {
    int i = blockIdx.x * blockDim.x + threadIdx.x;
    if (i < n) out[i] = (bf16_t)in[i];
}

// fp32 [K,C] row-major -> bf16 column-major Wt[c*K + k]
__global__ __launch_bounds__(256)
void k_wT2bf(const float* __restrict__ W, bf16_t* __restrict__ Wt, int K, int C) {
    int i = blockIdx.x * blockDim.x + threadIdx.x;
    if (i >= K * C) return;
    int k = i / C, c = i % C;
    Wt[(size_t)c * K + k] = (bf16_t)W[i];
}

// ---------------------------------------------------------------------------
// C[M,N] = A[M,K] @ B[K,N]
//   A: bf16 row-major (lda=K); Bt: bf16 COLUMN-major (Bt[n*K+k]); C: f32.
// Block = 256 threads = 8 waves; block tile = 256x64 rows; wave tile = 32x64
// (eight 16x16 WMMA accumulators, 2 A fragments / 4 B fragments per K-step).
// Full B panel [64 cols x K] staged in LDS once (<= 64KB), then the K loop
// runs without any barriers.
__global__ __launch_bounds__(256)
void k_gemm_bf16_wmma(const bf16_t* __restrict__ A, const bf16_t* __restrict__ Bt,
                      float* __restrict__ C, int M, int N, int K) {
    __shared__ bf16_t ldsB[64 * KMAX];           // [col][k], 64 KB max

    const int tid  = threadIdx.x;
    const int wave = tid >> 5;
    const int lane = tid & 31;
    const int l16  = lane & 15;
    const int half = lane >> 4;                  // 0: lanes 0-15, 1: lanes 16-31

    const int rowBase = blockIdx.x * 256 + wave * 32;
    const int colBase = blockIdx.y * 64;

    // ---- stage entire B panel into LDS (one barrier total) ----
    {
        const int chunks = (64 * K) / 8;         // 8 bf16 per 16-B chunk
        const int perCol = K / 8;
        for (int idx = tid; idx < chunks; idx += 256) {
            int c  = idx / perCol;
            int kk = (idx % perCol) * 8;
            *(bf16x8*)(&ldsB[c * K + kk]) =
                *(const bf16x8*)(Bt + (size_t)(colBase + c) * K + kk);
        }
    }
    __syncthreads();

    int aRow0 = rowBase + l16;       if (aRow0 >= M) aRow0 = M - 1;
    int aRow1 = rowBase + 16 + l16;  if (aRow1 >= M) aRow1 = M - 1;
    const bf16_t* aBase0 = A + (size_t)aRow0 * K;
    const bf16_t* aBase1 = A + (size_t)aRow1 * K;

    f32x8 acc[8];
    #pragma unroll
    for (int i = 0; i < 8; ++i) acc[i] = f32x8{};

    for (int k0 = 0; k0 < K; k0 += 32) {
        // Prefetch next A K-slices into near caches (will be used next iter).
        __builtin_prefetch(aBase0 + k0 + 32, 0, 3);
        __builtin_prefetch(aBase1 + k0 + 32, 0, 3);

        // A fragments per ISA 16-bit 16x32 layout:
        //   lanes 0-15  -> K = {k0..k0+7,  k0+16..k0+23}
        //   lanes 16-31 -> K = {k0+8..+15, k0+24..+31}
        union { bf16x16 v; bf16x8 h[2]; } af0, af1;
        {
            const bf16_t* ap0 = aBase0 + k0 + half * 8;
            af0.h[0] = *(const bf16x8*)ap0;
            af0.h[1] = *(const bf16x8*)(ap0 + 16);
            const bf16_t* ap1 = aBase1 + k0 + half * 8;
            af1.h[0] = *(const bf16x8*)ap1;
            af1.h[1] = *(const bf16x8*)(ap1 + 16);
        }

        // Four 32x16 B fragments from LDS; each feeds 2 WMMAs.
        #pragma unroll
        for (int t = 0; t < 4; ++t) {
            union { bf16x16 v; bf16x8 h[2]; } bu;
            const bf16_t* bp = &ldsB[(t * 16 + l16) * K + k0 + half * 8];
            bu.h[0] = *(const bf16x8*)bp;
            bu.h[1] = *(const bf16x8*)(bp + 16);
            bf16x16 bf = bu.v;
            acc[t]     = __builtin_amdgcn_wmma_f32_16x16x32_bf16(
                             false, af0.v, false, bf, (short)0, acc[t],     false, false);
            acc[4 + t] = __builtin_amdgcn_wmma_f32_16x16x32_bf16(
                             false, af1.v, false, bf, (short)0, acc[4 + t], false, false);
        }
    }

    // C layout: lanes 0-15 -> rows +0..7 (VGPR r), lanes 16-31 -> rows +8..15.
    #pragma unroll
    for (int rg = 0; rg < 2; ++rg) {
        #pragma unroll
        for (int r = 0; r < 8; ++r) {
            int row = rowBase + rg * 16 + half * 8 + r;
            if (row < M) {
                float* cp = C + (size_t)row * N + colBase + l16;
                cp[0]  = acc[rg * 4 + 0][r];
                cp[16] = acc[rg * 4 + 1][r];
                cp[32] = acc[rg * 4 + 2][r];
                cp[48] = acc[rg * 4 + 3][r];
            }
        }
    }
}

// ---------------------------------------------------------------------------
// Attention logits: s_src[n,h] = <h[n,h,:], a_src[h,:]>, same for dst.
__global__ __launch_bounds__(256)
void k_scores(const float* __restrict__ h, const float* __restrict__ as,
              const float* __restrict__ ad, float* __restrict__ ssrc,
              float* __restrict__ sdst, int n, int heads) {
    int i = blockIdx.x * blockDim.x + threadIdx.x;
    if (i >= n * heads) return;
    int node = i / heads, hd = i % heads;
    const float* hp = h + (size_t)node * heads * DHH + hd * DHH;
    const float* ap = as + hd * DHH;
    const float* bp = ad + hd * DHH;
    float s1 = 0.f, s2 = 0.f;
    #pragma unroll 8
    for (int d = 0; d < DHH; ++d) { float v = hp[d]; s1 += v * ap[d]; s2 += v * bp[d]; }
    ssrc[i] = s1; sdst[i] = s2;
}

__global__ __launch_bounds__(256)
void k_zero(float* __restrict__ p, long n) {
    long i = (long)blockIdx.x * blockDim.x + threadIdx.x;
    if (i < n) p[i] = 0.f;
}

__global__ __launch_bounds__(256)
void k_init_softmax(float* __restrict__ m, float* __restrict__ denom, int n) {
    int i = blockIdx.x * blockDim.x + threadIdx.x;
    if (i < n) { m[i] = -INFINITY; denom[i] = 0.f; }
}

__device__ __forceinline__ void atomicMaxF(float* addr, float v) {
    if (v >= 0.f) atomicMax((int*)addr, __float_as_int(v));
    else          atomicMin((unsigned int*)addr, __float_as_uint(v));
}

__device__ __forceinline__ void edge_sd(const int* ei, int idx, int& s, int& d) {
    if (idx < EE) { s = ei[idx]; d = ei[EE + idx]; }
    else          { s = d = idx - EE; }           // appended self-loops
}

// Pass 1: segment max of leaky_relu(s_src[src]+s_dst[dst]) per dst node.
__global__ __launch_bounds__(256)
void k_edge_max(const int* __restrict__ ei, const float* __restrict__ ssrc,
                const float* __restrict__ sdst, float* __restrict__ m, int heads) {
    int idx = blockIdx.x * blockDim.x + threadIdx.x;
    if (idx >= ETOT) return;
    int s, d; edge_sd(ei, idx, s, d);
    for (int h = 0; h < heads; ++h) {
        float v = ssrc[s * heads + h] + sdst[d * heads + h];
        v = v > 0.f ? v : 0.2f * v;
        atomicMaxF(&m[d * heads + h], v);
    }
}

// Pass 2: p = exp(e - m[dst]); store per-edge p; segment-sum denominator.
__global__ __launch_bounds__(256)
void k_edge_p(const int* __restrict__ ei, const float* __restrict__ ssrc,
              const float* __restrict__ sdst, const float* __restrict__ m,
              float* __restrict__ ew, float* __restrict__ denom, int heads) {
    int idx = blockIdx.x * blockDim.x + threadIdx.x;
    if (idx >= ETOT) return;
    int s, d; edge_sd(ei, idx, s, d);
    for (int h = 0; h < heads; ++h) {
        float v = ssrc[s * heads + h] + sdst[d * heads + h];
        v = v > 0.f ? v : 0.2f * v;
        float p = __expf(v - m[d * heads + h]);
        ew[(size_t)idx * heads + h] = p;
        atomicAdd(&denom[d * heads + h], p);
    }
}

// Pass 3: agg[dst] += h[src] * alpha   (alpha = p/denom), per feature.
__global__ __launch_bounds__(256)
void k_edge_agg(const int* __restrict__ ei, const float* __restrict__ h,
                const float* __restrict__ ew, const float* __restrict__ denom,
                float* __restrict__ agg, int heads, int C) {
    int idx = blockIdx.x;
    if (idx >= ETOT) return;
    int s, d; edge_sd(ei, idx, s, d);
    for (int f = threadIdx.x; f < C; f += blockDim.x) {
        int hd = f >> 6;
        float alpha = ew[(size_t)idx * heads + hd] / (denom[d * heads + hd] + 1e-16f);
        atomicAdd(&agg[(size_t)d * C + f], h[(size_t)s * C + f] * alpha);
    }
}

// ---------------------------------------------------------------------------
// BatchNorm column stats over (x + bias): partial sums per block, atomics out.
__global__ __launch_bounds__(256)
void k_col_stats(const float* __restrict__ x, const float* __restrict__ bias,
                 float* __restrict__ sum, float* __restrict__ sq,
                 int n, int C, int rowsPerBlock) {
    int r0 = blockIdx.x * rowsPerBlock;
    int r1 = min(n, r0 + rowsPerBlock);
    for (int c = threadIdx.x; c < C; c += blockDim.x) {
        float b = bias[c], s = 0.f, q = 0.f;
        for (int r = r0; r < r1; ++r) {
            float v = x[(size_t)r * C + c] + b;
            s += v; q += v * v;
        }
        atomicAdd(&sum[c], s);
        atomicAdd(&sq[c], q);
    }
}

// out = elu(bn(x + bias)) [+ residual]
__global__ __launch_bounds__(256)
void k_bn_elu(const float* __restrict__ x, const float* __restrict__ bias,
              const float* __restrict__ sum, const float* __restrict__ sq,
              const float* __restrict__ g, const float* __restrict__ b,
              const float* __restrict__ residual, float* __restrict__ out,
              int n, int C) {
    long i = (long)blockIdx.x * blockDim.x + threadIdx.x;
    if (i >= (long)n * C) return;
    int c = (int)(i % C);
    float inv  = 1.0f / (float)n;
    float mean = sum[c] * inv;
    float var  = sq[c] * inv - mean * mean;
    float v = (x[i] + bias[c] - mean) * rsqrtf(var + 1e-5f) * g[c] + b[c];
    v = v > 0.f ? v : expm1f(v);                 // ELU
    if (residual) v += residual[i];
    out[i] = v;
}

// Row LayerNorm over 64 features, in place.
__global__ __launch_bounds__(256)
void k_layernorm(float* __restrict__ x, const float* __restrict__ g,
                 const float* __restrict__ b, int n) {
    int r = blockIdx.x * blockDim.x + threadIdx.x;
    if (r >= n) return;
    float* p = x + (size_t)r * DHH;
    float mu = 0.f;
    for (int d = 0; d < DHH; ++d) mu += p[d];
    mu *= (1.0f / DHH);
    float var = 0.f;
    for (int d = 0; d < DHH; ++d) { float t = p[d] - mu; var += t * t; }
    var *= (1.0f / DHH);
    float rstd = rsqrtf(var + 1e-5f);
    for (int d = 0; d < DHH; ++d) p[d] = (p[d] - mu) * rstd * g[d] + b[d];
}

// Global mean pool: per-graph sums + counts.
__global__ __launch_bounds__(256)
void k_pool(const float* __restrict__ x, const int* __restrict__ batch,
            float* __restrict__ sums, float* __restrict__ cnt, int n) {
    long i = (long)blockIdx.x * blockDim.x + threadIdx.x;
    if (i >= (long)n * DHH) return;
    int r = (int)(i >> 6), c = (int)(i & 63);
    int gidx = batch[r];
    atomicAdd(&sums[gidx * DHH + c], x[i]);
    if (c == 0) atomicAdd(&cnt[gidx], 1.0f);
}

// Head: pooled mean -> fc1 (elu) -> fc2 -> log_softmax.  One thread per graph.
__global__ __launch_bounds__(32)
void k_head(const float* __restrict__ sums, const float* __restrict__ cnt,
            const float* __restrict__ w1, const float* __restrict__ b1,
            const float* __restrict__ w2, const float* __restrict__ b2,
            float* __restrict__ out) {
    int gi = blockIdx.x * blockDim.x + threadIdx.x;
    if (gi >= GG) return;
    float inv = 1.0f / fmaxf(cnt[gi], 1.0f);
    float pooled[DHH];
    for (int d = 0; d < DHH; ++d) pooled[d] = sums[gi * DHH + d] * inv;
    float hid[32];
    for (int j = 0; j < 32; ++j) {
        float s = b1[j];
        for (int d = 0; d < DHH; ++d) s += pooled[d] * w1[d * 32 + j];
        hid[j] = s > 0.f ? s : expm1f(s);
    }
    float l0 = b2[0], l1 = b2[1];
    for (int j = 0; j < 32; ++j) { l0 += hid[j] * w2[j * 2]; l1 += hid[j] * w2[j * 2 + 1]; }
    float mx  = fmaxf(l0, l1);
    float lse = mx + logf(__expf(l0 - mx) + __expf(l1 - mx));
    out[gi * 2]     = l0 - lse;
    out[gi * 2 + 1] = l1 - lse;
}

// ---------------------------------------------------------------------------
extern "C" void kernel_launch(void* const* d_in, const int* in_sizes, int n_in,
                              void* d_out, int out_size, void* d_ws, size_t ws_size,
                              hipStream_t stream) {
    (void)in_sizes; (void)n_in; (void)out_size; (void)ws_size;

    const float* x      = (const float*)d_in[0];
    const int*   ei     = (const int*)  d_in[1];
    const int*   batch  = (const int*)  d_in[2];
    const float* W0     = (const float*)d_in[3];
    const float* as0    = (const float*)d_in[4];
    const float* ad0    = (const float*)d_in[5];
    const float* b0     = (const float*)d_in[6];
    const float* W1     = (const float*)d_in[7];
    const float* as1    = (const float*)d_in[8];
    const float* ad1    = (const float*)d_in[9];
    const float* b1     = (const float*)d_in[10];
    const float* W2     = (const float*)d_in[11];
    const float* as2    = (const float*)d_in[12];
    const float* ad2    = (const float*)d_in[13];
    const float* b2     = (const float*)d_in[14];
    const float* bn0_g  = (const float*)d_in[15];
    const float* bn0_b  = (const float*)d_in[16];
    const float* bn1_g  = (const float*)d_in[17];
    const float* bn1_b  = (const float*)d_in[18];
    const float* bn2_g  = (const float*)d_in[19];
    const float* bn2_b  = (const float*)d_in[20];
    const float* ln_g   = (const float*)d_in[21];
    const float* ln_b   = (const float*)d_in[22];
    const float* fc1_w  = (const float*)d_in[23];
    const float* fc1_b  = (const float*)d_in[24];
    const float* fc2_w  = (const float*)d_in[25];
    const float* fc2_b  = (const float*)d_in[26];
    float* out = (float*)d_out;

    // ---- workspace carve-up (256-B aligned) ----
    char* ws = (char*)d_ws;
    size_t off = 0;
    auto alloc = [&](size_t bytes) -> char* {
        char* p = ws + off;
        off = (off + bytes + 255) & ~(size_t)255;
        return p;
    };
    bf16_t* abf  = (bf16_t*)alloc((size_t)NN * C512 * sizeof(bf16_t)); // GEMM A (bf16)
    bf16_t* wt   = (bf16_t*)alloc((size_t)C512 * C512 * sizeof(bf16_t)); // col-major W
    float* hbuf  = (float*)alloc((size_t)NN * C512 * sizeof(float));   // projection
    float* agg   = (float*)alloc((size_t)NN * C512 * sizeof(float));   // aggregation
    float* act1  = (float*)alloc((size_t)NN * C512 * sizeof(float));   // x1
    float* act2  = (float*)alloc((size_t)NN * C512 * sizeof(float));   // x2
    float* x3    = (float*)alloc((size_t)NN * DHH * sizeof(float));    // x3
    float* ssrc  = (float*)alloc((size_t)NN * HH * sizeof(float));
    float* sdst  = (float*)alloc((size_t)NN * HH * sizeof(float));
    float* mmax  = (float*)alloc((size_t)NN * HH * sizeof(float));
    float* denom = (float*)alloc((size_t)NN * HH * sizeof(float));
    float* ew    = (float*)alloc((size_t)ETOT * HH * sizeof(float));
    float* csum  = (float*)alloc((size_t)C512 * sizeof(float));
    float* csq   = (float*)alloc((size_t)C512 * sizeof(float));
    float* psum  = (float*)alloc((size_t)GG * DHH * sizeof(float));
    float* pcnt  = (float*)alloc((size_t)GG * sizeof(float));

    const int B = 256;
    auto blocks = [](long n, int b) { return (int)((n + b - 1) / b); };
    const int rowsPerBlk = 128;
    const int statBlocks = blocks(NN, rowsPerBlk);

    // One GAT layer: proj (WMMA) -> scores -> segment softmax -> aggregate -> BN+ELU.
    auto gat_layer = [&](const float* xin, int K, const float* W, int C, int heads,
                         const float* a_s, const float* a_d, const float* bias,
                         const float* bng, const float* bnb,
                         const float* residual, float* xout) {
        // bf16 inputs for WMMA
        k_f2bf <<<blocks((long)NN * K, B), B, 0, stream>>>(xin, abf, NN * K);
        k_wT2bf<<<blocks((long)K * C, B), B, 0, stream>>>(W, wt, K, C);
        // projection: hbuf[N,C] = xin @ W  (block tile 256 rows x 64 cols)
        dim3 gg(blocks(NN, 256), C / 64);
        k_gemm_bf16_wmma<<<gg, B, 0, stream>>>(abf, wt, hbuf, NN, C, K);
        // attention logits
        k_scores<<<blocks((long)NN * heads, B), B, 0, stream>>>(hbuf, a_s, a_d,
                                                               ssrc, sdst, NN, heads);
        // segment softmax + aggregation
        k_init_softmax<<<blocks((long)NN * heads, B), B, 0, stream>>>(mmax, denom, NN * heads);
        k_zero<<<blocks((long)NN * C, B), B, 0, stream>>>(agg, (long)NN * C);
        k_edge_max<<<blocks(ETOT, B), B, 0, stream>>>(ei, ssrc, sdst, mmax, heads);
        k_edge_p  <<<blocks(ETOT, B), B, 0, stream>>>(ei, ssrc, sdst, mmax, ew, denom, heads);
        k_edge_agg<<<ETOT, B, 0, stream>>>(ei, hbuf, ew, denom, agg, heads, C);
        // BN + ELU (+ residual)
        k_zero<<<blocks(2L * C, B), B, 0, stream>>>(csum, C);
        k_zero<<<blocks(2L * C, B), B, 0, stream>>>(csq, C);
        k_col_stats<<<statBlocks, B, 0, stream>>>(agg, bias, csum, csq, NN, C, rowsPerBlk);
        k_bn_elu<<<blocks((long)NN * C, B), B, 0, stream>>>(agg, bias, csum, csq,
                                                            bng, bnb, residual, xout, NN, C);
    };

    // layer 0: 256 -> 8x64, no residual
    gat_layer(x,    DIN,  W0, C512, HH, as0, ad0, b0, bn0_g, bn0_b, nullptr, act1);
    // layer 1: 512 -> 8x64, residual x1
    gat_layer(act1, C512, W1, C512, HH, as1, ad1, b1, bn1_g, bn1_b, act1,    act2);
    // layer 2: 512 -> 1x64, no residual
    gat_layer(act2, C512, W2, DHH,  1,  as2, ad2, b2, bn2_g, bn2_b, nullptr, x3);

    // LayerNorm, mean pool, MLP head, log_softmax
    k_layernorm<<<blocks(NN, B), B, 0, stream>>>(x3, ln_g, ln_b, NN);
    k_zero<<<blocks((long)GG * DHH, B), B, 0, stream>>>(psum, (long)GG * DHH);
    k_zero<<<1, B, 0, stream>>>(pcnt, GG);
    k_pool<<<blocks((long)NN * DHH, B), B, 0, stream>>>(x3, batch, psum, pcnt, NN);
    k_head<<<1, 32, 0, stream>>>(psum, pcnt, fc1_w, fc1_b, fc2_w, fc2_b, out);
}
